// RelativeScaledDotProductAttention_1778116460898
// MI455X (gfx1250) — matrix operand order repaired and checked by reading, not compiled, not run on hardware
//
#include <hip/hip_runtime.h>

typedef __attribute__((ext_vector_type(16))) _Float16 v16h;
typedef __attribute__((ext_vector_type(8)))  float    v8f;

#define B_   4
#define L_   1024
#define H_   8
#define DK_  32
#define DV_  32
#define ITILE 16
#define SSTRIDE 1028   // padded LDS row stride (floats) -> conflict-free banks

__launch_bounds__(256)
__global__ void rel_attn_kernel(const float* __restrict__ q,
                                const float* __restrict__ k,
                                const float* __restrict__ v,
                                const float* __restrict__ a_k,
                                float* __restrict__ out,    // [B,H,L,DV]
                                float* __restrict__ attn)   // [B,H,L,L]
{
    __shared__ float Slds[ITILE * SSTRIDE];   // 64.2 KB score tile (f32)
    __shared__ float Klds[ITILE * DK_];       // 2 KB K tile
    __shared__ float rowMax[ITILE];
    __shared__ float rowInv[ITILE];

    const int bh = blockIdx.x;        // fast axis: 32 blocks share same a_k i-tile -> L2 reuse
    const int b  = bh / H_;
    const int h  = bh % H_;
    const int i0 = blockIdx.y * ITILE;
    const int t  = threadIdx.x;

    // ---- load K tile (16 x 32) into LDS ----
    for (int e = t; e < ITILE * DK_; e += 256) {
        const int il = e >> 5, d = e & 31;
        Klds[e] = k[((size_t)b * L_ + (i0 + il)) * (H_ * DK_) + h * DK_ + d];
    }
    __syncthreads();

    const int il = t >> 4;     // 0..15: i row within tile
    const int jl = t & 15;     // 0..15: j lane

    float kreg[DK_];
#pragma unroll
    for (int d = 0; d < DK_; ++d) kreg[d] = Klds[il * DK_ + d];

    const float invT = 0.17677669529663687f;  // 1/sqrt(32)

    // ---- phase 1: scores S[i,j] = (sum_d k*q*a)/T  (elementwise a_k -> VALU f32) ----
    for (int jc = 0; jc < L_ / 16; ++jc) {
        const int j = jc * 16 + jl;
        const float4* A4 = (const float4*)(a_k + ((size_t)(i0 + il) * L_ + j) * DK_);
        const float4* Q4 = (const float4*)(q + ((size_t)b * L_ + j) * (H_ * DK_) + h * DK_);
        __builtin_prefetch((const char*)A4 + 16 * DK_ * 4, 0, 1);  // next jc chunk of a_k
        float s0 = 0.f, s1 = 0.f;
#pragma unroll
        for (int dd = 0; dd < 8; ++dd) {
            const float4 av = A4[dd];
            const float4 qv = Q4[dd];
            s0 = fmaf(kreg[4 * dd + 0] * av.x, qv.x, s0);
            s1 = fmaf(kreg[4 * dd + 1] * av.y, qv.y, s1);
            s0 = fmaf(kreg[4 * dd + 2] * av.z, qv.z, s0);
            s1 = fmaf(kreg[4 * dd + 3] * av.w, qv.w, s1);
        }
        Slds[il * SSTRIDE + j] = (s0 + s1) * invT;
    }
    __syncthreads();

    // ---- softmax stats: 16 lanes per row, wave32-safe shuffle reductions ----
    float m = -3.4e38f;
    for (int j = jl; j < L_; j += 16) m = fmaxf(m, Slds[il * SSTRIDE + j]);
#pragma unroll
    for (int off = 8; off > 0; off >>= 1) m = fmaxf(m, __shfl_xor(m, off, 16));
    float z = 0.f;
    for (int j = jl; j < L_; j += 16) z += expf(Slds[il * SSTRIDE + j] - m);
#pragma unroll
    for (int off = 8; off > 0; off >>= 1) z += __shfl_xor(z, off, 16);
    if (jl == 0) { rowMax[il] = m; rowInv[il] = 1.f / z; }
    __syncthreads();

    // ---- write normalized attn rows (non-temporal: don't evict a_k from L2) ----
    {
        const float mi = rowMax[il], zi = rowInv[il];
        float* arow = attn + ((size_t)bh * L_ + (i0 + il)) * L_;
        for (int j = jl; j < L_; j += 16) {
            const float p = expf(Slds[il * SSTRIDE + j] - mi) * zi;
            __builtin_nontemporal_store(p, arow + j);
        }
    }

    // ---- phase 2: out(16 x 32) = P(16 x 1024) @ V(1024 x 32) via WMMA f16->f32 ----
    const int wave  = t >> 5;
    const int lane  = t & 31;
    const int row   = lane & 15;                 // A: M index; B: N index
    const int jbaseA = (lane < 16) ? 0 : 8;      // ISA 16-bit A 16x32 lane layout
    const int kbaseB = (lane < 16) ? 0 : 8;      // ISA 16-bit B 32x16 lane layout

    v8f c0 = {};
    v8f c1 = {};

    for (int kk = 0; kk < 4; ++kk) {
        const int jt    = wave + kk * 8;         // 32 j-tiles of 32 across 8 waves
        const int jbase = jt * 32;

        // A fragment: attention probabilities, recomputed from LDS scores
        const float mi = rowMax[row], zi = rowInv[row];
        v16h a;
#pragma unroll
        for (int e = 0; e < 8; ++e) {
            const int jloc = jbaseA + e;          // K = 0..7 (or 8..15)
            a[e] = (_Float16)(expf(Slds[row * SSTRIDE + jbase + jloc] - mi) * zi);
        }
#pragma unroll
        for (int e = 8; e < 16; ++e) {
            const int jloc = jbaseA + 16 + (e - 8);  // K = 16..23 (or 24..31)
            a[e] = (_Float16)(expf(Slds[row * SSTRIDE + jbase + jloc] - mi) * zi);
        }

        // B fragments: V tile (32j x 32dv), split into two 32x16 N-halves
        v16h b0, b1;
#pragma unroll
        for (int e = 0; e < 16; ++e) {
            const int jj = jbase + kbaseB + e + ((e >= 8) ? 8 : 0);    // K (=j) index
            const float* vrow = v + ((size_t)b * L_ + jj) * (H_ * DV_) + h * DV_;
            b0[e] = (_Float16)vrow[row];          // dv = 0..15  (coalesced per half-wave)
            b1[e] = (_Float16)vrow[row + 16];     // dv = 16..31
        }

        c0 = __builtin_amdgcn_wmma_f32_16x16x32_f16(false, a, false, b0,
                                                    (short)0, c0, false, false);
        c1 = __builtin_amdgcn_wmma_f32_16x16x32_f16(false, a, false, b1,
                                                    (short)0, c1, false, false);
    }

    __syncthreads();   // everyone done reading Slds -> safe to reuse as reduction scratch

    float* red = Slds; // 4096 floats needed, fits easily
#pragma unroll
    for (int r = 0; r < 8; ++r) {
        red[((wave * 2 + 0) * 8 + r) * 32 + lane] = c0[r];
        red[((wave * 2 + 1) * 8 + r) * 32 + lane] = c1[r];
    }
    __syncthreads();

    // cross-wave reduce + store out; C layout: VGPR r, lanes 0-15 -> M=r, 16-31 -> M=r+8
    for (int e = t; e < ITILE * DV_; e += 256) {
        const int i  = e >> 5;      // 0..15
        const int dv = e & 31;      // 0..31
        const int f  = dv >> 4;     // which C fragment
        const int n  = dv & 15;
        const int r  = i & 7;
        const int ln = ((i >= 8) ? 16 : 0) + n;
        float acc = 0.f;
#pragma unroll
        for (int w = 0; w < 8; ++w)
            acc += red[((w * 2 + f) * 8 + r) * 32 + ln];
        out[((size_t)bh * L_ + (i0 + i)) * DV_ + dv] = acc;
    }
}

extern "C" void kernel_launch(void* const* d_in, const int* in_sizes, int n_in,
                              void* d_out, int out_size, void* d_ws, size_t ws_size,
                              hipStream_t stream) {
    const float* q   = (const float*)d_in[0];
    const float* k   = (const float*)d_in[1];
    const float* v   = (const float*)d_in[2];
    const float* a_k = (const float*)d_in[3];
    // d_in[4..6] are d_k, d_v, n_head scalars; compile-time constants here.

    float* out  = (float*)d_out;
    float* attn = out + (size_t)B_ * H_ * L_ * DV_;   // tuple: out first, then attn

    dim3 grid(B_ * H_, L_ / ITILE);   // bh fast -> concurrent blocks share a_k tile in L2
    rel_attn_kernel<<<grid, 256, 0, stream>>>(q, k, v, a_k, out, attn);
}
